// RecurrentCUBALIFCell_50053548868255
// MI455X (gfx1250) — matrix operand order repaired and checked by reading, not compiled, not run on hardware
//
#include <hip/hip_runtime.h>

#define BATCH  4096
#define INDIM  2048
#define HIDDEN 2048
#define LDA    2048

#define BM 128
#define BN 128
#define BK 32
#define PITCH 40                 // bf16 elements per LDS row (padded, 80B = 16B aligned)
#define KT1    (INDIM / BK)      // 64
#define KT_TOT ((INDIM + HIDDEN) / BK)  // 128

typedef __bf16 v16bf __attribute__((ext_vector_type(16)));
typedef float  v8f   __attribute__((ext_vector_type(8)));

union FragU { uint4 u[2]; v16bf v; };

// gfx1250 assembles v_cvt_pk_bf16_f32 (verified round 3).
static __device__ __forceinline__ unsigned int pack2_bf16(float a, float b) {
  unsigned int r;
  asm("v_cvt_pk_bf16_f32 %0, %1, %2" : "=v"(r) : "v"(a), "v"(b));
  return r;
}

static __device__ __forceinline__ float selc(float4 v, int j) {
  float r = v.x;
  r = (j == 1) ? v.y : r;
  r = (j == 2) ? v.z : r;
  r = (j == 3) ? v.w : r;
  return r;
}

// Manual workgroup barrier for the GEMM pipeline.
// __syncthreads() carries a workgroup fence that conservatively drains
// LOADcnt (s_wait_loadcnt 0x0 before s_barrier_signal, seen in rounds 3-6),
// defeating the cross-iteration register prefetch. The LDS double-buffer
// protocol only needs DScnt==0 before signaling (LDS stores visible + LDS
// reads retired; DScnt tracks both). Global loads target private VGPRs and
// may legally stay in flight across the barrier.
static __device__ __forceinline__ void pipe_barrier() {
  asm volatile("s_wait_dscnt 0x0\n\t"
               "s_barrier_signal -1\n\t"
               "s_barrier_wait -1" ::: "memory");
#if __has_builtin(__builtin_amdgcn_sched_barrier)
  __builtin_amdgcn_sched_barrier(0);   // keep pipeline regions intact
#endif
}

__global__ __launch_bounds__(256)
void lif_fused(const float* __restrict__ X,    const float* __restrict__ S,
               const float* __restrict__ preC, const float* __restrict__ preV,
               const float* __restrict__ preTh,const float* __restrict__ mask,
               const float* __restrict__ Wpsp, const float* __restrict__ bpsp,
               const float* __restrict__ Wrec, const float* __restrict__ brec,
               float* __restrict__ out, int nseg)
{
  __shared__ __align__(16) unsigned short As[2][BM * PITCH];  // [m][k] bf16
  __shared__ __align__(16) unsigned short Bs[2][BN * PITCH];  // [n][k] bf16 (W transposed)

  const int tid  = threadIdx.x;
  const int lane = tid & 31;
  const int wv   = tid >> 5;      // 8 waves
  const int wm   = wv >> 1;       // 0..3  (rows, 32 each)
  const int wn   = wv & 1;        // 0..1  (cols, 64 each)
  const int row0 = blockIdx.y * BM;
  const int col0 = blockIdx.x * BN;

  // global-load coordinates
  const int a_m  = tid >> 3;      // + it*32 -> A tile row
  const int a_kq = tid & 7;       // float4 within BK
  const int b_kb = 4 * wv;        // + it   -> W tile row (k)
  const int b_nq = lane;          // float4 within BN
  const int rot  = (lane >> 2) & 3;

  // loop-invariant per-thread element offsets (only uniform kb advances)
  int aoff[4], woff[4];
  #pragma unroll
  for (int it = 0; it < 4; ++it) {
    aoff[it] = (row0 + it * 32 + a_m) * LDA + 4 * a_kq;
    woff[it] = (b_kb + it) * HIDDEN + col0 + 4 * b_nq;
  }

  v8f acc[2][4];
  #pragma unroll
  for (int i = 0; i < 2; ++i)
    #pragma unroll
    for (int j = 0; j < 4; ++j)
      #pragma unroll
      for (int e = 0; e < 8; ++e) acc[i][j][e] = 0.0f;

  float4 pa0[4], pb0[4], pa1[4], pb1[4];   // two prefetch register sets

  auto gload = [&](float4 (&pa)[4], float4 (&pb)[4],
                   const float* __restrict__ Ap, const float* __restrict__ Wp, int kb) {
    const int kbH = kb * HIDDEN;
    #pragma unroll
    for (int it = 0; it < 4; ++it) {
      pa[it] = *(const float4*)&Ap[(size_t)(aoff[it] + kb)];
      pb[it] = *(const float4*)&Wp[(size_t)(woff[it] + kbH)];
    }
  };

  auto lstore = [&](int buf, const float4 (&pa)[4], const float4 (&pb)[4]) {
    #pragma unroll
    for (int it = 0; it < 4; ++it) {
      uint2 pk;
      pk.x = pack2_bf16(pa[it].x, pa[it].y);
      pk.y = pack2_bf16(pa[it].z, pa[it].w);
      *(uint2*)&As[buf][(it * 32 + a_m) * PITCH + 4 * a_kq] = pk;
    }
    #pragma unroll
    for (int j = 0; j < 4; ++j) {
      int jj = (j + rot) & 3;     // rotated order -> bank-spread transpose scatter
      uint2 pk;
      pk.x = pack2_bf16(selc(pb[0], jj), selc(pb[1], jj));
      pk.y = pack2_bf16(selc(pb[2], jj), selc(pb[3], jj));
      *(uint2*)&Bs[buf][(4 * b_nq + jj) * PITCH + b_kb] = pk;
    }
  };

  auto compute = [&](int buf) {
    const int l16 = lane & 15, hf = lane >> 4;
    FragU af[2], bfr[4];
    #pragma unroll
    for (int mt = 0; mt < 2; ++mt) {
      const unsigned short* p = &As[buf][(wm * 32 + mt * 16 + l16) * PITCH + hf * 16];
      af[mt].u[0] = *(const uint4*)p;          // k: half*16 + 0..7
      af[mt].u[1] = *(const uint4*)(p + 8);    // k: half*16 + 8..15
    }
    #pragma unroll
    for (int nt = 0; nt < 4; ++nt) {
      const unsigned short* p = &Bs[buf][(wn * 64 + nt * 16 + l16) * PITCH + hf * 16];
      bfr[nt].u[0] = *(const uint4*)p;         // k: half*16 + 0..7
      bfr[nt].u[1] = *(const uint4*)(p + 8);   // k: half*16 + 8..15
    }
    #pragma unroll
    for (int mt = 0; mt < 2; ++mt)
      #pragma unroll
      for (int nt = 0; nt < 4; ++nt)
        acc[mt][nt] = __builtin_amdgcn_wmma_f32_16x16x32_bf16(
            false, af[mt].v, false, bfr[nt].v, (short)0, acc[mt][nt], false, false);
  };

  // ---- prologue: tiles 0 and 1 in flight, tile 0 in LDS buf0 ----
  gload(pa0, pb0, X, Wpsp, 0);          // tile 0 -> set0
  gload(pa1, pb1, X, Wpsp, BK);         // tile 1 -> set1
  lstore(0, pa0, pb0);                  // tile 0 -> buf0
  pipe_barrier();

  // ---- phase 1: X @ W_psp ; iterations kt = 0..61 (tiles gloaded: 2..63) ----
  for (int kk = 0; kk < 31; ++kk) {
    const int kt = 2 * kk;
    lstore(1, pa1, pb1);                        // tile kt+1 -> buf1
    gload(pa0, pb0, X, Wpsp, (kt + 2) * BK);    // tile kt+2 -> set0
    compute(0);                                 // tile kt
    pipe_barrier();
    lstore(0, pa0, pb0);                        // tile kt+2 -> buf0
    gload(pa1, pb1, X, Wpsp, (kt + 3) * BK);    // tile kt+3 -> set1
    compute(1);                                 // tile kt+1
    pipe_barrier();
  }

  // ---- boundary: kt = 62, 63 (gload phase-2 tiles 64, 65) ----
  lstore(1, pa1, pb1);                  // tile 63 -> buf1
  gload(pa0, pb0, S, Wrec, 0);          // tile 64 -> set0
  compute(0);                           // tile 62
  pipe_barrier();
  lstore(0, pa0, pb0);                  // tile 64 -> buf0
  gload(pa1, pb1, S, Wrec, BK);         // tile 65 -> set1
  compute(1);                           // tile 63
  pipe_barrier();

  // ---- phase 2: S @ W_rec ; iterations kt = 64..125 (tiles gloaded: 66..127) ----
  for (int kk = 0; kk < 31; ++kk) {
    const int kt = 64 + 2 * kk;
    lstore(1, pa1, pb1);
    gload(pa0, pb0, S, Wrec, (kt + 2 - KT1) * BK);
    compute(0);
    pipe_barrier();
    lstore(0, pa0, pb0);
    gload(pa1, pb1, S, Wrec, (kt + 3 - KT1) * BK);
    compute(1);
    pipe_barrier();
  }

  // ---- tail: kt = 126, 127 ----
  lstore(1, pa1, pb1);                  // tile 127 -> buf1
  compute(0);                           // tile 126
  pipe_barrier();
  compute(1);                           // tile 127

  // ---- fused LIF epilogue on C fragments ----
  const size_t NTOT = (size_t)BATCH * HIDDEN;
  const int l16 = lane & 15, hf = lane >> 4;
  #pragma unroll
  for (int nt = 0; nt < 4; ++nt) {
    const int col  = col0 + wn * 64 + nt * 16 + l16;
    const float bsum = bpsp[col] + brec[col];
    #pragma unroll
    for (int mt = 0; mt < 2; ++mt) {
      const int rbase = row0 + wm * 32 + mt * 16 + hf * 8;
      #pragma unroll
      for (int r = 0; r < 8; ++r) {
        const size_t idx = (size_t)(rbase + r) * HIDDEN + col;
        float c = fmaf(0.5f, __builtin_nontemporal_load(&preC[idx]),
                       acc[mt][nt][r] + bsum);
        c *= __builtin_nontemporal_load(&mask[idx]);
        const float ps  = S[idx];
        const float vlt = fmaf(0.75f * __builtin_nontemporal_load(&preV[idx]),
                               1.0f - ps, c);
        const float th  = __builtin_nontemporal_load(&preTh[idx]);
        const bool  sp  = vlt > th;
        const float o   = sp ? 1.0f : 0.0f;
        const float nth = sp ? (th + 0.05f) : fmaxf(th * 0.9f, 1.0f);
        for (int s = 0; s <= nseg - 4; ++s)                 // output copies
          __builtin_nontemporal_store(o, &out[(size_t)s * NTOT + idx]);
        __builtin_nontemporal_store(c,   &out[(size_t)(nseg - 3) * NTOT + idx]);
        __builtin_nontemporal_store(vlt, &out[(size_t)(nseg - 2) * NTOT + idx]);
        __builtin_nontemporal_store(nth, &out[(size_t)(nseg - 1) * NTOT + idx]);
      }
    }
  }
}

extern "C" void kernel_launch(void* const* d_in, const int* in_sizes, int n_in,
                              void* d_out, int out_size, void* d_ws, size_t ws_size,
                              hipStream_t stream) {
  (void)in_sizes; (void)n_in; (void)d_ws; (void)ws_size;
  const float* X     = (const float*)d_in[0];
  const float* S     = (const float*)d_in[1];
  const float* preC  = (const float*)d_in[2];
  const float* preV  = (const float*)d_in[3];
  const float* preTh = (const float*)d_in[4];
  const float* mask  = (const float*)d_in[5];
  const float* Wpsp  = (const float*)d_in[6];
  const float* bpsp  = (const float*)d_in[7];
  const float* Wrec  = (const float*)d_in[8];
  const float* brec  = (const float*)d_in[9];
  float* out = (float*)d_out;

  const long long NTOT = (long long)BATCH * HIDDEN;
  int nseg = (int)((long long)out_size / NTOT);   // expect 5: out,out,current,volt,vth
  if (nseg < 4) nseg = 4;

  dim3 grid(HIDDEN / BN, BATCH / BM);             // 16 x 32 = 512 workgroups
  lif_fused<<<grid, 256, 0, stream>>>(X, S, preC, preV, preTh, mask,
                                      Wpsp, bpsp, Wrec, brec, out, nseg);
}
